// MultiHeadAttention_35665408426664
// MI455X (gfx1250) — matrix operand order repaired and checked by reading, not compiled, not run on hardware
//
#include <hip/hip_runtime.h>

// ---------------------------------------------------------------------------
// MHA for MI455X (gfx1250, wave32): bf16 WMMA everywhere, f32 accumulate.
// B=2, L=2048, D=1024, H=16, Hd=64.
// Round 2 additions:
//   * Tensor Data Mover (tensor_load_to_lds) stages the non-transposed tiles
//     (GEMM A 128x32, attention K 64x64): wave0 issues one DMA per tile,
//     s_wait_tensorcnt 0 + workgroup barrier publishes it. Double-buffered so
//     the DMA of tile n+1 flies under the WMMA block of tile n.
//   * DPP row_ror softmax reductions (VALU pipe), 64-key attention steps.
// ---------------------------------------------------------------------------

#define SEQ_L   2048
#define D_MODEL 1024
#define N_HEADS 16
#define HEAD_D  64
#define QKV_W   (3 * D_MODEL)
#define TOKENS  (2 * SEQ_L)

typedef unsigned short u16;
typedef __attribute__((ext_vector_type(16))) __bf16 v16bf;
typedef __attribute__((ext_vector_type(8)))  float  v8f;
typedef __attribute__((ext_vector_type(4)))  unsigned int u32x4;
typedef __attribute__((ext_vector_type(4)))  int          i32x4;
typedef __attribute__((ext_vector_type(8)))  int          i32x8;

#if defined(__has_builtin)
#if __has_builtin(__builtin_amdgcn_tensor_load_to_lds) && \
    __has_builtin(__builtin_amdgcn_s_wait_tensorcnt)
#define HAVE_TDM 1
#endif
#endif

union Frag  { uint4 q[2]; v16bf v; };   // 32B = 16 bf16 (one WMMA A/B operand)
union Pack8 { u16 s[16]; uint4 q[2]; };

__device__ __forceinline__ u16 f2bf(float f) {      // RNE float -> bf16 bits
  unsigned u = __float_as_uint(f);
  u += 0x7FFFu + ((u >> 16) & 1u);
  return (u16)(u >> 16);
}

__device__ __forceinline__ v8f vzero() {
  v8f z = {0.f, 0.f, 0.f, 0.f, 0.f, 0.f, 0.f, 0.f};
  return z;
}

__device__ __forceinline__ v8f wmma_bf16(const Frag& a, const Frag& b, v8f c) {
  // D = A(16x32 bf16) * B(32x16 bf16) + C(16x16 f32)
  return __builtin_amdgcn_wmma_f32_16x16x32_bf16(
      /*neg_a=*/false, a.v, /*neg_b=*/false, b.v,
      /*c_mod=*/(short)0, c, /*reuse_a=*/false, /*reuse_b=*/false);
}

#ifdef HAVE_TDM
// ---- TDM: DMA a 2D bf16 tile (tile_w x tile_h elems, row stride in elems)
// from global memory into contiguous LDS. Wave-level op (EXEC ignored);
// caller gates to one wave and waits TENSORcnt before the publishing barrier.
// D# packing per CDNA5 ISA ch.8 (group0 128b, group1 256b; groups 2/3 unused
// for 2D tensors).
__device__ __forceinline__ void tdm_load_2d(unsigned lds_addr, const void* gp,
                                            unsigned tile_w, unsigned tile_h,
                                            unsigned stride_elems) {
  unsigned long long ga = (unsigned long long)gp;
  u32x4 g0 = {
    1u,                                               // count=1, user mode
    lds_addr,                                         // LDS byte address
    (unsigned)ga,                                     // global_addr[31:0]
    (unsigned)((ga >> 32) & 0x01FFFFFFu) | 0x80000000u // addr[56:32] | type=2
  };
  i32x8 g1 = {
    (int)0x00010000u,                                  // data_size=1 (2 bytes)
    (int)((tile_w & 0xFFFFu) << 16),                   // tensor_dim0[15:0]
    (int)((tile_w >> 16) | ((tile_h & 0xFFFFu) << 16)),// dim0 hi | tensor_dim1 lo
    (int)((tile_h >> 16) | (tile_w << 16)),            // dim1 hi | tile_dim0
    (int)(tile_h & 0xFFFFu),                           // tile_dim1 (tile_dim2=0)
    (int)stride_elems,                                 // tensor_dim0_stride lo
    0, 0
  };
  i32x4 z4 = {0, 0, 0, 0};
#if __clang_major__ >= 23
  i32x8 z8 = {0, 0, 0, 0, 0, 0, 0, 0};
  __builtin_amdgcn_tensor_load_to_lds(g0, g1, z4, z4, z8, 0);
#else
  __builtin_amdgcn_tensor_load_to_lds(g0, g1, z4, z4, 0);
#endif
}
__device__ __forceinline__ unsigned lds_addr_of(const void* p) {
  return (unsigned)(unsigned long long)p;   // generic addr low 32 = LDS offset
}
#endif

// ---- 16-lane row reductions on the VALU pipe (DPP row_ror 8/4/2/1). -------
#define ROW_ROR(x, n) \
  __builtin_amdgcn_update_dpp(0, (x), 0x120 | (n), 0xF, 0xF, true)

__device__ __forceinline__ float row16_allmax(float v) {
  v = fmaxf(v, __int_as_float(ROW_ROR(__float_as_int(v), 8)));
  v = fmaxf(v, __int_as_float(ROW_ROR(__float_as_int(v), 4)));
  v = fmaxf(v, __int_as_float(ROW_ROR(__float_as_int(v), 2)));
  v = fmaxf(v, __int_as_float(ROW_ROR(__float_as_int(v), 1)));
  return v;
}
__device__ __forceinline__ float row16_allsum(float v) {
  v += __int_as_float(ROW_ROR(__float_as_int(v), 8));
  v += __int_as_float(ROW_ROR(__float_as_int(v), 4));
  v += __int_as_float(ROW_ROR(__float_as_int(v), 2));
  v += __int_as_float(ROW_ROR(__float_as_int(v), 1));
  return v;
}

// ---------------------------------------------------------------------------
// fp32 -> bf16 conversion, 4 elements/thread
// ---------------------------------------------------------------------------
__global__ void __launch_bounds__(256)
cvt_f32_bf16(const float* __restrict__ src, u16* __restrict__ dst, int n4) {
  int i = blockIdx.x * 256 + threadIdx.x;
  if (i >= n4) return;
  float4 f = ((const float4*)src)[i];
  union { u16 s[4]; uint2 q; } p;
  p.s[0] = f2bf(f.x); p.s[1] = f2bf(f.y); p.s[2] = f2bf(f.z); p.s[3] = f2bf(f.w);
  ((uint2*)dst)[i] = p.q;
}

// ---------------------------------------------------------------------------
// Tiled bf16 GEMM + bias: C[M,N] = A[M,K] * B[K,N] + bias[N]
// Block tile 128x128, K-step 32, 8 waves, each wave computes 32x64.
// A tile staged by TDM (row-major, contiguous LDS); B tile staged per-lane
// transposed to [n][k]; both fragment types are contiguous ds_load_b128 pairs.
// ---------------------------------------------------------------------------
template <bool OUT_BF16>
__global__ void __launch_bounds__(256)
gemm_bf16_tc(const u16* __restrict__ A, const u16* __restrict__ B,
             const float* __restrict__ bias, void* __restrict__ Cout,
             int M, int N, int K) {
  __shared__ __align__(16) u16 As[2][128 * 32];   // [m][k]
  __shared__ __align__(16) u16 Bs[2][128 * 32];   // [n][k]  (transposed)

  const int tid   = threadIdx.x;
  const int wid   = tid >> 5;
  const int lane  = tid & 31;
  const int lrow  = lane & 15;
  const int lhalf = lane >> 4;
  const int m_w   = wid >> 1;        // 0..3
  const int n_w   = wid & 1;         // 0..1
  const int m0    = blockIdx.y * 128;
  const int n0    = blockIdx.x * 128;

  v8f acc[2][4];
#pragma unroll
  for (int mi = 0; mi < 2; ++mi)
#pragma unroll
    for (int ni = 0; ni < 4; ++ni) acc[mi][ni] = vzero();

  const int bn = tid & 127, bk = (tid >> 7) << 4; // B: 128 cols x 2 k-halves
#ifndef HAVE_TDM
  const int arow = tid >> 1, aseg = tid & 1;      // A: 128 rows x 2 halves
#endif

  auto stageA = [&](u16* Ad, int k0) {
#ifdef HAVE_TDM
    if (wid == 0)
      tdm_load_2d(lds_addr_of(Ad), A + (size_t)m0 * K + k0, 32, 128, K);
#else
    const u16* ag = A + (size_t)(m0 + arow) * K + k0 + aseg * 16;
    *(uint4*)&Ad[arow * 32 + aseg * 16]     = *(const uint4*)ag;
    *(uint4*)&Ad[arow * 32 + aseg * 16 + 8] = *(const uint4*)(ag + 8);
#endif
  };
  auto stageB = [&](u16* Bd, int k0) {
    Pack8 pb;
#pragma unroll
    for (int j = 0; j < 16; ++j)
      pb.s[j] = B[(size_t)(k0 + bk + j) * N + n0 + bn];
    *(uint4*)&Bd[bn * 32 + bk]     = pb.q[0];
    *(uint4*)&Bd[bn * 32 + bk + 8] = pb.q[1];
  };

  stageA(As[0], 0);                            // prologue
  stageB(Bs[0], 0);

  const int nsteps = K >> 5;
  for (int it = 0; it < nsteps; ++it) {
    const int cur = it & 1;
#ifdef HAVE_TDM
    if (wid == 0) __builtin_amdgcn_s_wait_tensorcnt(0);
#endif
    __syncthreads();                           // stage(cur) done; nxt free
    if (it + 1 < nsteps) {
      stageA(As[cur ^ 1], (it + 1) * 32);
      stageB(Bs[cur ^ 1], (it + 1) * 32);
    }
    if (it + 2 < nsteps)
      __builtin_prefetch(&B[(size_t)((it + 2) * 32 + bk) * N + n0 + bn], 0, 0);

    const u16* Ab = As[cur];
    const u16* Bb = Bs[cur];
    Frag af[2], bfr[4];
#pragma unroll
    for (int mi = 0; mi < 2; ++mi) {
      const u16* p = &Ab[(m_w * 32 + mi * 16 + lrow) * 32];
      af[mi].q[0] = *(const uint4*)(p + lhalf * 8);        // K = h*8 .. +7
      af[mi].q[1] = *(const uint4*)(p + 16 + lhalf * 8);   // K = 16+h*8 .. +7
    }
#pragma unroll
    for (int ni = 0; ni < 4; ++ni) {
      const u16* p = &Bb[(n_w * 64 + ni * 16 + lrow) * 32 + lhalf * 8];
      bfr[ni].q[0] = *(const uint4*)(p);                   // K = h*8 .. +7
      bfr[ni].q[1] = *(const uint4*)(p + 16);              // K = 16+h*8 .. +7
    }
#pragma unroll
    for (int mi = 0; mi < 2; ++mi)
#pragma unroll
      for (int ni = 0; ni < 4; ++ni)
        acc[mi][ni] = wmma_bf16(af[mi], bfr[ni], acc[mi][ni]);
  }

  // ---- epilogue: +bias, store (C layout: VGPR i -> row i+8*half, col lrow) --
#pragma unroll
  for (int mi = 0; mi < 2; ++mi)
#pragma unroll
    for (int ni = 0; ni < 4; ++ni) {
      const int col  = n0 + n_w * 64 + ni * 16 + lrow;
      const float bv = bias[col];
      const int rb   = m0 + m_w * 32 + mi * 16 + 8 * lhalf;
#pragma unroll
      for (int i = 0; i < 8; ++i) {
        float v = acc[mi][ni][i] + bv;
        size_t idx = (size_t)(rb + i) * N + col;
        if (OUT_BF16) ((u16*)Cout)[idx]  = f2bf(v);
        else          ((float*)Cout)[idx] = v;
      }
    }
}

// ---------------------------------------------------------------------------
// Fused flash attention (bf16 WMMA, f32 accumulate, online softmax).
// Grid: (L/128, B*H). Block 256 = 8 waves; each wave owns 16 q rows.
// K-step = 64 keys: K tile [64 tok][64 d] staged by TDM, V tile transposed
// per-lane to [64 d][64 tok]; both double-buffered.
// ---------------------------------------------------------------------------
__global__ void __launch_bounds__(256)
attn_fused(const u16* __restrict__ qkv, u16* __restrict__ o) {
  __shared__ __align__(16) u16 Ks[2][64 * 64];     // [tok][d]
  __shared__ __align__(16) u16 Vs[2][64 * 64];     // [d][tok]
  __shared__ __align__(16) u16 Ps[8][16 * 64];     // per-wave P tile

  const int tid   = threadIdx.x;
  const int wid   = tid >> 5;
  const int lane  = tid & 31;
  const int lrow  = lane & 15;
  const int lhalf = lane >> 4;
  const int qblk  = blockIdx.x;
  const int b     = blockIdx.y >> 4;
  const int h     = blockIdx.y & 15;
  const int tq0   = b * SEQ_L + qblk * 128 + wid * 16;

  // ---- Q fragments: row = lrow, two 16x32 A-frags over Hd=64 ----
  const u16* qg = qkv + (size_t)(tq0 + lrow) * QKV_W + h * HEAD_D;
  Frag qf[2];
#pragma unroll
  for (int j = 0; j < 2; ++j) {
    qf[j].q[0] = *(const uint4*)(qg + j * 32 + lhalf * 8);
    qf[j].q[1] = *(const uint4*)(qg + j * 32 + 16 + lhalf * 8);
  }

  v8f oacc[4];
#pragma unroll
  for (int ni = 0; ni < 4; ++ni) oacc[ni] = vzero();
  float m_run[8], l_run[8];
#pragma unroll
  for (int i = 0; i < 8; ++i) { m_run[i] = -3.0e38f; l_run[i] = 0.f; }

  const int vd = tid & 63, vtg = tid >> 6;     // V stage: d, token-group of 16
  const size_t kbase = (size_t)(b * SEQ_L) * QKV_W + D_MODEL + h * HEAD_D;
  const size_t vbase = (size_t)(b * SEQ_L) * QKV_W + 2 * D_MODEL + h * HEAD_D;
#ifndef HAVE_TDM
  const int ktok = tid >> 2, kpart = tid & 3;  // K stage: token, 16-elem chunk
#endif

  auto stageK = [&](u16* Kd, int t0) {
#ifdef HAVE_TDM
    if (wid == 0)
      tdm_load_2d(lds_addr_of(Kd), qkv + kbase + (size_t)t0 * QKV_W,
                  64, 64, QKV_W);
#else
    const u16* kg = qkv + kbase + (size_t)(t0 + ktok) * QKV_W + kpart * 16;
    *(uint4*)&Kd[ktok * 64 + kpart * 16]     = *(const uint4*)kg;
    *(uint4*)&Kd[ktok * 64 + kpart * 16 + 8] = *(const uint4*)(kg + 8);
#endif
  };
  auto stageV = [&](u16* Vd, int t0) {
    Pack8 pv;
#pragma unroll
    for (int j = 0; j < 16; ++j)
      pv.s[j] = qkv[vbase + (size_t)(t0 + vtg * 16 + j) * QKV_W + vd];
    *(uint4*)&Vd[vd * 64 + vtg * 16]     = pv.q[0];
    *(uint4*)&Vd[vd * 64 + vtg * 16 + 8] = pv.q[1];
  };

  stageK(Ks[0], 0);                             // prologue
  stageV(Vs[0], 0);

  const int nsteps = SEQ_L / 64;
  for (int it = 0; it < nsteps; ++it) {
    const int cur = it & 1;
#ifdef HAVE_TDM
    if (wid == 0) __builtin_amdgcn_s_wait_tensorcnt(0);
#endif
    __syncthreads();                            // stage(cur) done; nxt free
    if (it + 1 < nsteps) {
      stageK(Ks[cur ^ 1], (it + 1) * 64);
      stageV(Vs[cur ^ 1], (it + 1) * 64);
    }

    const u16* Kb = Ks[cur];
    const u16* Vb = Vs[cur];

    // ---- S = Q @ K^T : four 16-key tiles, contraction over d=64 ----
    v8f s[4];
#pragma unroll
    for (int nt = 0; nt < 4; ++nt) {
      s[nt] = vzero();
#pragma unroll
      for (int j = 0; j < 2; ++j) {
        Frag kf;
        const u16* p = &Kb[(nt * 16 + lrow) * 64 + j * 32 + lhalf * 8];
        kf.q[0] = *(const uint4*)p;
        kf.q[1] = *(const uint4*)(p + 16);
        s[nt] = wmma_bf16(qf[j], kf, s[nt]);
      }
    }

    // ---- online softmax per accumulator row (row = i + 8*lhalf) ----
    u16* prow = &Ps[wid][0];
#pragma unroll
    for (int i = 0; i < 8; ++i) {
      float sv[4];
#pragma unroll
      for (int nt = 0; nt < 4; ++nt) sv[nt] = s[nt][i] * 0.125f; // 1/sqrt(Hd)
      float mx = row16_allmax(fmaxf(fmaxf(sv[0], sv[1]), fmaxf(sv[2], sv[3])));
      float m_new = fmaxf(m_run[i], mx);
      float alpha = __expf(m_run[i] - m_new);
      float p0 = __expf(sv[0] - m_new);
      float p1 = __expf(sv[1] - m_new);
      float p2 = __expf(sv[2] - m_new);
      float p3 = __expf(sv[3] - m_new);
      float rs = row16_allsum((p0 + p1) + (p2 + p3));
      l_run[i] = l_run[i] * alpha + rs;
      m_run[i] = m_new;
#pragma unroll
      for (int ni = 0; ni < 4; ++ni) oacc[ni][i] *= alpha;
      const int r = (i + 8 * lhalf) * 64 + lrow;
      prow[r]      = f2bf(p0);
      prow[r + 16] = f2bf(p1);
      prow[r + 32] = f2bf(p2);
      prow[r + 48] = f2bf(p3);
    }

    // ---- O += P @ V : two kv-halves x four d-tiles ----
#pragma unroll
    for (int j = 0; j < 2; ++j) {
      Frag pf;
      const u16* pp = &Ps[wid][lrow * 64 + j * 32 + lhalf * 8];
      pf.q[0] = *(const uint4*)pp;
      pf.q[1] = *(const uint4*)(pp + 16);
#pragma unroll
      for (int ni = 0; ni < 4; ++ni) {
        Frag vf;
        const u16* p = &Vb[(ni * 16 + lrow) * 64 + j * 32 + lhalf * 8];
        vf.q[0] = *(const uint4*)p;
        vf.q[1] = *(const uint4*)(p + 16);
        oacc[ni] = wmma_bf16(pf, vf, oacc[ni]);
      }
    }
  }

  // ---- finalize: divide by row sum, emit bf16 ----
#pragma unroll
  for (int i = 0; i < 8; ++i) {
    float inv = 1.0f / l_run[i];
    int row = tq0 + i + 8 * lhalf;
#pragma unroll
    for (int ni = 0; ni < 4; ++ni) {
      int col = h * HEAD_D + ni * 16 + lrow;
      o[(size_t)row * D_MODEL + col] = f2bf(oacc[ni][i] * inv);
    }
  }
}

// ---------------------------------------------------------------------------
extern "C" void kernel_launch(void* const* d_in, const int* in_sizes, int n_in,
                              void* d_out, int out_size, void* d_ws, size_t ws_size,
                              hipStream_t stream) {
  const float* x     = (const float*)d_in[0];
  const float* w_qkv = (const float*)d_in[1];
  const float* b_qkv = (const float*)d_in[2];
  const float* w_out = (const float*)d_in[3];
  const float* b_out = (const float*)d_in[4];
  float* out = (float*)d_out;

  // workspace layout (bf16 as u16): 48 MiB total
  u16* xb   = (u16*)d_ws;                          // [4096 x 1024]
  u16* wqb  = xb   + (size_t)TOKENS * D_MODEL;     // [1024 x 3072]
  u16* wob  = wqb  + (size_t)D_MODEL * QKV_W;      // [1024 x 1024]
  u16* qkvb = wob  + (size_t)D_MODEL * D_MODEL;    // [4096 x 3072]
  u16* ob   = qkvb + (size_t)TOKENS * QKV_W;       // [4096 x 1024]

  int n;
  n = TOKENS * D_MODEL / 4;
  cvt_f32_bf16<<<(n + 255) / 256, 256, 0, stream>>>(x, xb, n);
  n = D_MODEL * QKV_W / 4;
  cvt_f32_bf16<<<(n + 255) / 256, 256, 0, stream>>>(w_qkv, wqb, n);
  n = D_MODEL * D_MODEL / 4;
  cvt_f32_bf16<<<(n + 255) / 256, 256, 0, stream>>>(w_out, wob, n);

  // qkv = x @ w_qkv + b_qkv  -> bf16
  gemm_bf16_tc<true><<<dim3(QKV_W / 128, TOKENS / 128), 256, 0, stream>>>(
      xb, wqb, b_qkv, qkvb, TOKENS, QKV_W, D_MODEL);

  // fused attention -> o (bf16)
  attn_fused<<<dim3(SEQ_L / 128, 2 * N_HEADS), 256, 0, stream>>>(qkvb, ob);

  // out = o @ w_out + b_out -> f32
  gemm_bf16_tc<false><<<dim3(D_MODEL / 128, TOKENS / 128), 256, 0, stream>>>(
      ob, wob, b_out, out, TOKENS, D_MODEL, D_MODEL);
}
